// AttSplineConv_65962107732907
// MI455X (gfx1250) — compile-verified, matrix operand order checked by reading, not run on hardware
//
#include <hip/hip_runtime.h>
#include <hip/hip_bf16.h>
#include <math.h>

#define N_NODES 50000
#define N_EDGES 800000
#define KPROD   25
#define FEAT    64
#define NEG_SLOPE 0.2f

#define TILES      (N_NODES / 16)      // 3125 -- exact, no partial tiles
#define WT_STRIDE  72                  // padded LDS column stride (elements);
                                       // 72*2=144B keeps 16B alignment, kills
                                       // the 16-way bank conflict of stride 64

typedef __attribute__((ext_vector_type(16))) __bf16 v16bf;
typedef __attribute__((ext_vector_type(8)))  __bf16 v8bf;
typedef __attribute__((ext_vector_type(8)))  float  v8f;

// ---------------------------------------------------------------------------
// helpers
// ---------------------------------------------------------------------------

// A-fragment (16x32 bf16) per ISA 7.12.2: lane l holds row M=(l&15);
// element e -> K = kbase + (e&7) + (e>=8 ? 16 : 0), kbase = kc*32 + (l>=16)*8.
__device__ inline v16bf load_a_frag(const float* __restrict__ xrow, int kbase) {
  v16bf a;
#pragma unroll
  for (int e = 0; e < 8; ++e) {
    a[e]     = (__bf16)xrow[kbase + e];
    a[e + 8] = (__bf16)xrow[kbase + 16 + e];
  }
  return a;
}

// B-fragment (32x16 bf16) from LDS holding W transposed ([out][in], bf16):
// lane l holds col N=(l&15); same K pattern as A (symmetric layout).
__device__ inline v16bf load_b_frag(const __bf16* wt_col, int kbase) {
  const v8bf b0 = *(const v8bf*)(wt_col + kbase);        // K = kbase+0..7
  const v8bf b1 = *(const v8bf*)(wt_col + kbase + 16);   // K = kbase+16..23
  v16bf b;
#pragma unroll
  for (int e = 0; e < 8; ++e) { b[e] = b0[e]; b[e + 8] = b1[e]; }
  return b;
}

__device__ inline void atomicMaxF(float* addr, float v) {
  // ordered-int trick: works for any sign, init value must be -inf
  if (v >= 0.0f) atomicMax((int*)addr, __float_as_int(v));
  else           atomicMin((unsigned int*)addr, __float_as_uint(v));
}

// ---------------------------------------------------------------------------
// kernel 0: init accumulators
// ---------------------------------------------------------------------------
__global__ void k_init(float* __restrict__ mx, float* __restrict__ den,
                       float* __restrict__ agg) {
  int i = blockIdx.x * blockDim.x + threadIdx.x;
  if (i < N_NODES) { mx[i] = -INFINITY; den[i] = 0.0f; }
  if (i < N_NODES * FEAT) agg[i] = 0.0f;
}

// ---------------------------------------------------------------------------
// kernel 1: h_all[n,k,o] = (x @ weight[k]) in bf16 (160 MB -> fits 192MB L2)
// 8 waves/block, wave w owns node-tile (blockIdx*8+w); loop k staging W[k]^T
// in LDS as bf16; 16x16x32 bf16 WMMA, f32 accumulate. N_NODES%16==0 so the
// only guard is a wave-uniform tile bound (stores are straight-line).
// ---------------------------------------------------------------------------
__global__ void k_hall(const float* __restrict__ x,
                       const float* __restrict__ weight,
                       __bf16* __restrict__ h_all) {
  __shared__ __bf16 WT[FEAT * WT_STRIDE];   // WT[o*WT_STRIDE + i]

  const int lane = threadIdx.x & 31;
  const int wave = threadIdx.x >> 5;
  const int tile = blockIdx.x * 8 + wave;
  const bool active = (tile < TILES);       // wave-uniform
  const int m0   = tile * 16;
  const int half = lane >> 4;
  const int l15  = lane & 15;

  const int mrow  = m0 + l15;
  const int mload = active ? mrow : 0;
  const float* xrow = x + (size_t)mload * FEAT;

  v16bf afrag[2];
  afrag[0] = load_a_frag(xrow,      half * 8);
  afrag[1] = load_a_frag(xrow, 32 + half * 8);

  for (int k = 0; k < KPROD; ++k) {
    __syncthreads();
    for (int i = threadIdx.x; i < FEAT * FEAT; i += blockDim.x) {
      const int r = i >> 6, c = i & 63;           // r = input i, c = output o
      WT[c * WT_STRIDE + r] = (__bf16)weight[(size_t)k * (FEAT * FEAT) + i];
    }
    __syncthreads();
    if (!active) continue;

#pragma unroll
    for (int nc = 0; nc < 4; ++nc) {
      const int n = nc * 16 + l15;
      const __bf16* wt_col = &WT[n * WT_STRIDE];
      v8f c = {};
#pragma unroll
      for (int kc = 0; kc < 2; ++kc) {
        const v16bf b = load_b_frag(wt_col, kc * 32 + half * 8);
        c = __builtin_amdgcn_wmma_f32_16x16x32_bf16(
                false, afrag[kc], false, b, (short)0, c, false, false);
      }
      __bf16* outp = h_all + (size_t)(m0 + half * 8) * (KPROD * FEAT)
                           + (size_t)k * FEAT + n;
#pragma unroll
      for (int r = 0; r < 8; ++r)
        outp[(size_t)r * (KPROD * FEAT)] = (__bf16)c[r];
    }
  }
}

// ---------------------------------------------------------------------------
// kernel 2: root = x @ root_weight (f32 out), same WMMA structure, single k
// ---------------------------------------------------------------------------
__global__ void k_root(const float* __restrict__ x,
                       const float* __restrict__ rw,
                       float* __restrict__ root) {
  __shared__ __bf16 WT[FEAT * WT_STRIDE];

  const int lane = threadIdx.x & 31;
  const int wave = threadIdx.x >> 5;
  const int tile = blockIdx.x * 8 + wave;
  const bool active = (tile < TILES);
  const int m0   = tile * 16;
  const int half = lane >> 4;
  const int l15  = lane & 15;

  const int mload = active ? (m0 + l15) : 0;
  const float* xrow = x + (size_t)mload * FEAT;

  v16bf afrag[2];
  afrag[0] = load_a_frag(xrow,      half * 8);
  afrag[1] = load_a_frag(xrow, 32 + half * 8);

  for (int i = threadIdx.x; i < FEAT * FEAT; i += blockDim.x) {
    const int r = i >> 6, c = i & 63;
    WT[c * WT_STRIDE + r] = (__bf16)rw[i];
  }
  __syncthreads();
  if (!active) return;

#pragma unroll
  for (int nc = 0; nc < 4; ++nc) {
    const int n = nc * 16 + l15;
    const __bf16* wt_col = &WT[n * WT_STRIDE];
    v8f c = {};
#pragma unroll
    for (int kc = 0; kc < 2; ++kc) {
      const v16bf b = load_b_frag(wt_col, kc * 32 + half * 8);
      c = __builtin_amdgcn_wmma_f32_16x16x32_bf16(
              false, afrag[kc], false, b, (short)0, c, false, false);
    }
    float* outp = root + (size_t)(m0 + half * 8) * FEAT + n;
#pragma unroll
    for (int r = 0; r < 8; ++r)
      outp[(size_t)r * FEAT] = c[r];
  }
}

// ---------------------------------------------------------------------------
// kernel 3: rdot[n] = root[n,:] . att[0:64]
// ---------------------------------------------------------------------------
__global__ void k_rdot(const float* __restrict__ root,
                       const float* __restrict__ att,
                       float* __restrict__ rdot) {
  const int n = blockIdx.x * blockDim.x + threadIdx.x;
  if (n >= N_NODES) return;
  const float* r = root + (size_t)n * FEAT;
  float s = 0.0f;
#pragma unroll 8
  for (int o = 0; o < FEAT; ++o) s += r[o] * att[o];
  rdot[n] = s;
}

// ---------------------------------------------------------------------------
// kernel 4 (edge pass 1): wave32 per edge. spline basis, gather 4 L2-resident
// bf16 rows of h_all, msg = sum basis*h; alpha = leaky(rdot[dst]+msg.a_j);
// segment-max via ordered-int float atomic max.
// ---------------------------------------------------------------------------
__global__ void k_edge1(const int* __restrict__ ei,
                        const float* __restrict__ pseudo,
                        const __bf16* __restrict__ h_all,
                        const float* __restrict__ rdot,
                        const float* __restrict__ att,
                        __bf16* __restrict__ msgb,
                        float* __restrict__ alphaB,
                        float* __restrict__ mx) {
  const int gid  = blockIdx.x * blockDim.x + threadIdx.x;
  const int e    = gid >> 5;
  const int lane = gid & 31;
  if (e >= N_EDGES) return;

  const int src = ei[e];
  const int dst = ei[N_EDGES + e];

  const float p0 = pseudo[2 * e]     * 4.0f;   // (KS-1) = 4
  const float p1 = pseudo[2 * e + 1] * 4.0f;
  const int k00 = min(max((int)floorf(p0), 0), 3);
  const int k01 = min(max((int)floorf(p1), 0), 3);
  const float f0 = p0 - (float)k00;
  const float f1 = p1 - (float)k01;

  const __bf16* hb = h_all + (size_t)src * (KPROD * FEAT);
  float msg0 = 0.0f, msg1 = 0.0f;
#pragma unroll
  for (int s = 0; s < 4; ++s) {
    const int b0 = s >> 1, b1 = s & 1;                  // meshgrid 'ij' bit order
    const float bas = (b0 ? f0 : 1.0f - f0) * (b1 ? f1 : 1.0f - f1);
    const int idx = (k00 + b0) + 5 * (k01 + b1);        // strides [1,5]
    msg0 += bas * (float)hb[idx * FEAT + lane];
    msg1 += bas * (float)hb[idx * FEAT + lane + 32];
  }
  msgb[(size_t)e * FEAT + lane]      = (__bf16)msg0;
  msgb[(size_t)e * FEAT + lane + 32] = (__bf16)msg1;

  float part = msg0 * att[FEAT + lane] + msg1 * att[FEAT + lane + 32];
#pragma unroll
  for (int off = 16; off > 0; off >>= 1) part += __shfl_xor(part, off, 32);

  if (lane == 0) {
    float a = rdot[dst] + part;
    a = (a > 0.0f) ? a : NEG_SLOPE * a;
    alphaB[e] = a;
    atomicMaxF(mx + dst, a);
  }
}

// ---------------------------------------------------------------------------
// kernel 5 (edge pass 2): ex = exp(alpha - mx[dst]); den[dst] += ex
// ---------------------------------------------------------------------------
__global__ void k_edge2(const int* __restrict__ ei,
                        float* __restrict__ alphaB,
                        const float* __restrict__ mx,
                        float* __restrict__ den) {
  const int e = blockIdx.x * blockDim.x + threadIdx.x;
  if (e >= N_EDGES) return;
  const int dst = ei[N_EDGES + e];
  const float ex = __expf(alphaB[e] - mx[dst]);
  alphaB[e] = ex;
  atomicAdd(den + dst, ex);
}

// ---------------------------------------------------------------------------
// kernel 6 (edge pass 3): agg[dst,:] += (ex/den[dst]) * msg[e,:]
// ---------------------------------------------------------------------------
__global__ void k_edge3(const int* __restrict__ ei,
                        const float* __restrict__ exB,
                        const float* __restrict__ den,
                        const __bf16* __restrict__ msgb,
                        float* __restrict__ agg) {
  const int gid  = blockIdx.x * blockDim.x + threadIdx.x;
  const int e    = gid >> 5;
  const int lane = gid & 31;
  if (e >= N_EDGES) return;
  const int dst = ei[N_EDGES + e];
  const float coef = exB[e] / den[dst];
  atomicAdd(agg + (size_t)dst * FEAT + lane,
            coef * (float)msgb[(size_t)e * FEAT + lane]);
  atomicAdd(agg + (size_t)dst * FEAT + lane + 32,
            coef * (float)msgb[(size_t)e * FEAT + lane + 32]);
}

// ---------------------------------------------------------------------------
// kernel 7: out = agg + root + bias
// ---------------------------------------------------------------------------
__global__ void k_final(const float* __restrict__ agg,
                        const float* __restrict__ root,
                        const float* __restrict__ bias,
                        float* __restrict__ out) {
  const int i = blockIdx.x * blockDim.x + threadIdx.x;
  if (i >= N_NODES * FEAT) return;
  out[i] = agg[i] + root[i] + bias[i & 63];
}

// ---------------------------------------------------------------------------
// launch
// ---------------------------------------------------------------------------
extern "C" void kernel_launch(void* const* d_in, const int* in_sizes, int n_in,
                              void* d_out, int out_size, void* d_ws, size_t ws_size,
                              hipStream_t stream) {
  (void)in_sizes; (void)n_in; (void)out_size; (void)ws_size;

  const float* x      = (const float*)d_in[0];
  const int*   ei     = (const int*)d_in[1];
  const float* pseudo = (const float*)d_in[2];
  const float* weight = (const float*)d_in[3];
  const float* rootw  = (const float*)d_in[4];
  const float* att    = (const float*)d_in[5];
  const float* bias   = (const float*)d_in[6];
  float* out = (float*)d_out;

  char* base = (char*)d_ws;
  size_t off = 0;
  auto take = [&](size_t bytes) -> char* {
    char* p = base + off;
    off += (bytes + 255) & ~(size_t)255;
    return p;
  };
  __bf16* h_all = (__bf16*)take((size_t)N_NODES * KPROD * FEAT * 2);  // 160 MB
  __bf16* msgb  = (__bf16*)take((size_t)N_EDGES * FEAT * 2);          // 102 MB
  float*  root  = (float*)take((size_t)N_NODES * FEAT * 4);
  float*  agg   = (float*)take((size_t)N_NODES * FEAT * 4);
  float*  rdot  = (float*)take((size_t)N_NODES * 4);
  float*  alphaB= (float*)take((size_t)N_EDGES * 4);
  float*  mx    = (float*)take((size_t)N_NODES * 4);
  float*  den   = (float*)take((size_t)N_NODES * 4);

  const int gemmBlocks = (TILES + 7) / 8;       // 391 blocks x 8 waves

  hipLaunchKernelGGL(k_init,  dim3((N_NODES * FEAT + 255) / 256), dim3(256), 0, stream,
                     mx, den, agg);
  hipLaunchKernelGGL(k_hall,  dim3(gemmBlocks), dim3(256), 0, stream,
                     x, weight, h_all);
  hipLaunchKernelGGL(k_root,  dim3(gemmBlocks), dim3(256), 0, stream,
                     x, rootw, root);
  hipLaunchKernelGGL(k_rdot,  dim3((N_NODES + 255) / 256), dim3(256), 0, stream,
                     root, att, rdot);
  hipLaunchKernelGGL(k_edge1, dim3((N_EDGES * 32 + 255) / 256), dim3(256), 0, stream,
                     ei, pseudo, h_all, rdot, att, msgb, alphaB, mx);
  hipLaunchKernelGGL(k_edge2, dim3((N_EDGES + 255) / 256), dim3(256), 0, stream,
                     ei, alphaB, mx, den);
  hipLaunchKernelGGL(k_edge3, dim3((N_EDGES * 32 + 255) / 256), dim3(256), 0, stream,
                     ei, alphaB, den, msgb, agg);
  hipLaunchKernelGGL(k_final, dim3((N_NODES * FEAT + 255) / 256), dim3(256), 0, stream,
                     agg, root, bias, out);
}
